// GCGRU_20933670601173
// MI455X (gfx1250) — matrix operand order repaired and checked by reading, not compiled, not run on hardware
//
#include <hip/hip_runtime.h>
#include <hip/hip_bf16.h>

// ---------------------------------------------------------------------------
// GCGRU on MI455X (gfx1250): bf16 WMMA (16x16x32, fp32 accum) for all GEMMs,
// fp32 gate math / hidden state. Sequential scan over T with per-step fused
// GEMM+epilogue kernels. Wave tile 32x64 (2x4 frags) -> 8 WMMA per 6 frag
// loads per K-step.
// ---------------------------------------------------------------------------

typedef __bf16 bf16_t;
typedef __attribute__((ext_vector_type(16))) __bf16 v16bf;
typedef __attribute__((ext_vector_type(8)))  __bf16 v8bf;
typedef __attribute__((ext_vector_type(8)))  float  v8f;

#define B_    8
#define T_    64
#define N_    256
#define H_    256
#define DIN_  6
#define DOUT_ 9
#define G3_   768          // 3*H
#define BN_   (B_ * N_)    // 2048 flattened rows

// ---------------- WMMA fragment helpers (ISA 7.12.2 layouts) ----------------
// A-operand 16x32 bf16: lane L -> row = L%16, K in {base..base+7, base+16..base+23},
// base = (L/16)*8.  Source is row-major [M,K], K contiguous.
__device__ __forceinline__ v16bf load_fragA(const bf16_t* __restrict__ p, int ld, int lane) {
    const bf16_t* q = p + (lane & 15) * ld + ((lane >> 4) << 3);
    v8bf lo = *(const v8bf*)(q);        // K = base .. base+7
    v8bf hi = *(const v8bf*)(q + 16);   // K = base+16 .. base+23
    v16bf f;
#pragma unroll
    for (int e = 0; e < 8; ++e) { f[e] = lo[e]; f[e + 8] = hi[e]; }
    return f;
}

// B-operand 32x16 bf16: lane L -> col = L%16, K in {base..base+7, base+16..base+23}, base = (L/16)*8.
// Source is stored K-contiguous as Bt[N, K] (i.e. B transposed).
__device__ __forceinline__ v16bf load_fragB(const bf16_t* __restrict__ p, int ld, int lane) {
    const bf16_t* q = p + (lane & 15) * ld + ((lane >> 4) << 3);
    v8bf lo = *(const v8bf*)(q);
    v8bf hi = *(const v8bf*)(q + 16);
    v16bf f;
#pragma unroll
    for (int e = 0; e < 8; ++e) { f[e] = lo[e]; f[e + 8] = hi[e]; }
    return f;
}

__device__ __forceinline__ void zero_acc(v8f acc[2][4]) {
#pragma unroll
    for (int i = 0; i < 2; ++i)
#pragma unroll
        for (int j = 0; j < 4; ++j)
#pragma unroll
            for (int e = 0; e < 8; ++e) acc[i][j][e] = 0.0f;
}

// One wave computes a 32x64 output tile (2x4 fragments); WG = 4 waves = 64x128.
__device__ __forceinline__ void wmma_tile_32x64(const bf16_t* __restrict__ Atile, int lda,
                                                const bf16_t* __restrict__ Btile, int ldb,
                                                int K, int lane, v8f acc[2][4]) {
#pragma unroll 2
    for (int k = 0; k < K; k += 32) {
        v16bf a0 = load_fragA(Atile + k, lda, lane);
        v16bf a1 = load_fragA(Atile + 16 * lda + k, lda, lane);
        v16bf b0 = load_fragB(Btile + k, ldb, lane);
        v16bf b1 = load_fragB(Btile + 16 * ldb + k, ldb, lane);
        v16bf b2 = load_fragB(Btile + 32 * ldb + k, ldb, lane);
        v16bf b3 = load_fragB(Btile + 48 * ldb + k, ldb, lane);
        acc[0][0] = __builtin_amdgcn_wmma_f32_16x16x32_bf16(false, a0, false, b0, (short)0, acc[0][0], false, false);
        acc[0][1] = __builtin_amdgcn_wmma_f32_16x16x32_bf16(false, a0, false, b1, (short)0, acc[0][1], false, false);
        acc[0][2] = __builtin_amdgcn_wmma_f32_16x16x32_bf16(false, a0, false, b2, (short)0, acc[0][2], false, false);
        acc[0][3] = __builtin_amdgcn_wmma_f32_16x16x32_bf16(false, a0, false, b3, (short)0, acc[0][3], false, false);
        acc[1][0] = __builtin_amdgcn_wmma_f32_16x16x32_bf16(false, a1, false, b0, (short)0, acc[1][0], false, false);
        acc[1][1] = __builtin_amdgcn_wmma_f32_16x16x32_bf16(false, a1, false, b1, (short)0, acc[1][1], false, false);
        acc[1][2] = __builtin_amdgcn_wmma_f32_16x16x32_bf16(false, a1, false, b2, (short)0, acc[1][2], false, false);
        acc[1][3] = __builtin_amdgcn_wmma_f32_16x16x32_bf16(false, a1, false, b3, (short)0, acc[1][3], false, false);
    }
}

// WG tile 64x128: 4 waves arranged 2 (rows) x 2 (cols); wave tile 32x64.
struct TileCoord { int m0, n0, lane; };
__device__ __forceinline__ TileCoord tile_coord() {
    int wave = threadIdx.x >> 5;
    TileCoord tc;
    tc.lane = threadIdx.x & 31;
    tc.m0 = blockIdx.y * 64 + (wave >> 1) * 32;
    tc.n0 = blockIdx.x * 128 + (wave & 1) * 64;
    return tc;
}

__device__ __forceinline__ float sigmoid_f(float x) { return 1.0f / (1.0f + __expf(-x)); }

// ---------------- GEMM kernels -----------------------------------------------

// Batched Out[b] = Aadj(256x256) @ Ht[b]^T : grid (2,4,B), block 128.
// Aadj row-major [N,K=N] bf16; Ht[b] stored [col(j), K(m)] bf16 (i.e. h^T);
// Out[b] row-major [N, H] bf16.
__global__ void k_gemm_bf_out(const bf16_t* __restrict__ Aadj,
                              const bf16_t* __restrict__ Ht,
                              bf16_t* __restrict__ Out) {
    TileCoord tc = tile_coord();
    const bf16_t* Bz = Ht + (size_t)blockIdx.z * (H_ * N_);
    bf16_t* Oz = Out + (size_t)blockIdx.z * (N_ * H_);
    v8f acc[2][4]; zero_acc(acc);
    wmma_tile_32x64(Aadj + tc.m0 * N_, N_, Bz + tc.n0 * N_, N_, N_, tc.lane, acc);
    int hi = tc.lane >> 4, lo = tc.lane & 15;
#pragma unroll
    for (int i = 0; i < 2; ++i)
#pragma unroll
        for (int j = 0; j < 4; ++j) {
            int col  = tc.n0 + j * 16 + lo;
            int mrow = tc.m0 + i * 16 + hi * 8;
#pragma unroll
            for (int v = 0; v < 8; ++v)
                Oz[(mrow + v) * H_ + col] = (bf16_t)acc[i][j][v];
        }
}

// zr gates: S = Ah[2048,256] @ Wzrt^T (+gx), sigmoid.
// cols 0..255 -> z (fp32), cols 256..511 -> r; writes rht[b][j][n] = bf16(r*h).
// grid (4,32,1), block 128.
__global__ void k_gemm_zr(const bf16_t* __restrict__ Ah,
                          const bf16_t* __restrict__ Wzrt,   // [512,256]
                          const float*  __restrict__ gx,     // [2048,768]
                          const float*  __restrict__ hf,     // [2048,256]
                          float*        __restrict__ zbuf,   // [2048,256]
                          bf16_t*       __restrict__ rht) {  // [B][H][N]
    TileCoord tc = tile_coord();
    v8f acc[2][4]; zero_acc(acc);
    wmma_tile_32x64(Ah + tc.m0 * H_, H_, Wzrt + tc.n0 * H_, H_, H_, tc.lane, acc);
    int hi = tc.lane >> 4, lo = tc.lane & 15;
#pragma unroll
    for (int i = 0; i < 2; ++i)
#pragma unroll
        for (int j = 0; j < 4; ++j) {
            int col = tc.n0 + j * 16 + lo;
            int r0  = tc.m0 + i * 16 + hi * 8;
            if (col < 256) {
#pragma unroll
                for (int v = 0; v < 8; ++v) {
                    int r = r0 + v;
                    float g = acc[i][j][v] + gx[r * G3_ + col];
                    zbuf[r * H_ + col] = sigmoid_f(g);
                }
            } else {
                int jj = col - 256;
                v8bf pack;
#pragma unroll
                for (int v = 0; v < 8; ++v) {
                    int r = r0 + v;
                    float g = acc[i][j][v] + gx[r * G3_ + col];
                    float s = sigmoid_f(g);
                    pack[v] = (bf16_t)(s * hf[r * H_ + jj]);
                }
                int b = r0 >> 8, n0n = r0 & 255;   // 8 consecutive nodes -> one 16B store
                *(v8bf*)&rht[(size_t)b * (H_ * N_) + jj * N_ + n0n] = pack;
            }
        }
}

// candidate + GRU update: C = Arh[2048,256] @ Wct^T (+gx_c), tanh;
// h = z*h + (1-z)*c ; writes fp32 h and transposed bf16 h.
// grid (2,32,1), block 128.
__global__ void k_gemm_cupd(const bf16_t* __restrict__ Arh,
                            const bf16_t* __restrict__ Wct,   // [256,256]
                            const float*  __restrict__ gx,
                            const float*  __restrict__ zbuf,
                            float*        __restrict__ hf,
                            bf16_t*       __restrict__ ht) {  // [B][H][N]
    TileCoord tc = tile_coord();
    v8f acc[2][4]; zero_acc(acc);
    wmma_tile_32x64(Arh + tc.m0 * H_, H_, Wct + tc.n0 * H_, H_, H_, tc.lane, acc);
    int hi = tc.lane >> 4, lo = tc.lane & 15;
#pragma unroll
    for (int i = 0; i < 2; ++i)
#pragma unroll
        for (int j = 0; j < 4; ++j) {
            int col = tc.n0 + j * 16 + lo;
            int r0  = tc.m0 + i * 16 + hi * 8;
            v8bf pack;
#pragma unroll
            for (int v = 0; v < 8; ++v) {
                int r = r0 + v;
                float c = tanhf(acc[i][j][v] + gx[r * G3_ + 512 + col]);
                float z = zbuf[r * H_ + col];
                float hn = z * hf[r * H_ + col] + (1.0f - z) * c;
                hf[r * H_ + col] = hn;
                pack[v] = (bf16_t)hn;
            }
            int b = r0 >> 8, n0n = r0 & 255;
            *(v8bf*)&ht[(size_t)b * (H_ * N_) + col * N_ + n0n] = pack;
        }
}

// layer-1 input gates: gx = Axt[2048,256] @ Wxt^T + bias.  grid (6,32,1).
__global__ void k_gemm_gx(const bf16_t* __restrict__ Axt,
                          const bf16_t* __restrict__ Wxt,    // [768,256]
                          const float*  __restrict__ bias,   // [768]
                          float*        __restrict__ gx) {   // [2048,768]
    TileCoord tc = tile_coord();
    v8f acc[2][4]; zero_acc(acc);
    wmma_tile_32x64(Axt + tc.m0 * H_, H_, Wxt + tc.n0 * H_, H_, H_, tc.lane, acc);
    int hi = tc.lane >> 4, lo = tc.lane & 15;
#pragma unroll
    for (int i = 0; i < 2; ++i)
#pragma unroll
        for (int j = 0; j < 4; ++j) {
            int col = tc.n0 + j * 16 + lo;
            int r0  = tc.m0 + i * 16 + hi * 8;
            float bc = bias[col];
#pragma unroll
            for (int v = 0; v < 8; ++v)
                gx[(r0 + v) * G3_ + col] = acc[i][j][v] + bc;
        }
}

// ---------------- small VALU kernels -----------------------------------------

__global__ void k_cvt_bf(const float* __restrict__ src, bf16_t* __restrict__ dst, int n) {
    int i = blockIdx.x * blockDim.x + threadIdx.x;
    if (i < n) dst[i] = (bf16_t)src[i];
}

// W[rows,cols] fp32 -> Wt[cols,rows] bf16
__global__ void k_transpose_bf(const float* __restrict__ W, bf16_t* __restrict__ Wt,
                               int rows, int cols) {
    int i = blockIdx.x * blockDim.x + threadIdx.x;
    if (i < rows * cols) {
        int k = i / cols, c = i - k * cols;
        Wt[c * rows + k] = (bf16_t)W[i];
    }
}

__global__ void k_zero_f(float* __restrict__ p, int n) {
    int i = blockIdx.x * blockDim.x + threadIdx.x;
    if (i < n) p[i] = 0.0f;
}

// Ax0[b,n,d] = sum_m A[n,m] * x2d[b,t,m,d] * mask[b,t,m]   (DIN=6, tiny)
__global__ void k_ax0(const float* __restrict__ x2d, const float* __restrict__ mask,
                      const float* __restrict__ Aadj, float* __restrict__ Ax0, int t) {
    int i = blockIdx.x * blockDim.x + threadIdx.x;
    if (i >= B_ * N_ * DIN_) return;
    int d = i % DIN_;
    int n = (i / DIN_) % N_;
    int b = i / (DIN_ * N_);
    const float* xrow = x2d + (size_t)(b * T_ + t) * N_ * DIN_;
    const float* mrow = mask + (size_t)(b * T_ + t) * N_;
    const float* arow = Aadj + n * N_;
    float s = 0.0f;
    for (int m = 0; m < N_; ++m) s += arow[m] * xrow[m * DIN_ + d] * mrow[m];
    Ax0[(b * N_ + n) * DIN_ + d] = s;
}

// gx[r,c] = b0[c] + sum_{d<6} Ax0[r,d] * Wx0[d,c]
__global__ void k_gx0(const float* __restrict__ Ax0, const float* __restrict__ Wx0,
                      const float* __restrict__ b0, float* __restrict__ gx) {
    int i = blockIdx.x * blockDim.x + threadIdx.x;
    if (i >= BN_ * G3_) return;
    int c = i % G3_, r = i / G3_;
    const float* a = Ax0 + r * DIN_;
    float s = b0[c];
#pragma unroll
    for (int d = 0; d < DIN_; ++d) s += a[d] * Wx0[d * G3_ + c];
    gx[i] = s;
}

// out[b,t,n,o] = bout[o] + sum_j h1[b,n,j] * Wout[j,o]   (DOUT=9)
__global__ void k_out(const float* __restrict__ h1f, const float* __restrict__ Wout,
                      const float* __restrict__ bout, float* __restrict__ out, int t) {
    int i = blockIdx.x * blockDim.x + threadIdx.x;
    if (i >= B_ * N_ * DOUT_) return;
    int o = i % DOUT_;
    int n = (i / DOUT_) % N_;
    int b = i / (DOUT_ * N_);
    const float* h = h1f + (b * N_ + n) * H_;
    float s = bout[o];
    for (int j = 0; j < H_; ++j) s += h[j] * Wout[j * DOUT_ + o];
    out[((size_t)(b * T_ + t) * N_ + n) * DOUT_ + o] = s;
}

// ---------------- host launcher ----------------------------------------------

extern "C" void kernel_launch(void* const* d_in, const int* in_sizes, int n_in,
                              void* d_out, int out_size, void* d_ws, size_t ws_size,
                              hipStream_t stream) {
    (void)in_sizes; (void)n_in; (void)out_size; (void)ws_size;
    const float* x2d  = (const float*)d_in[0];
    const float* mask = (const float*)d_in[1];
    const float* sadj = (const float*)d_in[2];
    const float* Wx0  = (const float*)d_in[3];
    const float* Wh0  = (const float*)d_in[4];
    const float* b0   = (const float*)d_in[5];
    const float* Wx1  = (const float*)d_in[6];
    const float* Wh1  = (const float*)d_in[7];
    const float* b1   = (const float*)d_in[8];
    const float* Wout = (const float*)d_in[9];
    const float* bout = (const float*)d_in[10];
    float* out = (float*)d_out;

    char* ws = (char*)d_ws;
    size_t off = 0;
    auto take = [&](size_t bytes) -> char* {
        char* p = ws + off;
        off += (bytes + 255) & ~(size_t)255;
        return p;
    };
    bf16_t* A_bf   = (bf16_t*)take((size_t)N_ * N_ * 2);
    bf16_t* Wh0_t  = (bf16_t*)take((size_t)G3_ * H_ * 2);   // [768,256]; rows 0..511 = zr, 512.. = c
    bf16_t* Wh1_t  = (bf16_t*)take((size_t)G3_ * H_ * 2);
    bf16_t* Wx1_t  = (bf16_t*)take((size_t)G3_ * H_ * 2);
    float*  h0f    = (float*)take((size_t)BN_ * H_ * 4);    // h0f,h1f contiguous
    float*  h1f    = (float*)take((size_t)BN_ * H_ * 4);
    bf16_t* h0t    = (bf16_t*)take((size_t)B_ * H_ * N_ * 2); // h0t,h1t contiguous
    bf16_t* h1t    = (bf16_t*)take((size_t)B_ * H_ * N_ * 2);
    bf16_t* Ah_bf  = (bf16_t*)take((size_t)BN_ * H_ * 2);
    bf16_t* Axt_bf = (bf16_t*)take((size_t)BN_ * H_ * 2);
    bf16_t* rht    = (bf16_t*)take((size_t)B_ * H_ * N_ * 2);
    float*  zbuf   = (float*)take((size_t)BN_ * H_ * 4);
    float*  gx     = (float*)take((size_t)BN_ * G3_ * 4);
    float*  ax0    = (float*)take((size_t)BN_ * DIN_ * 4);

    dim3 blk(256), gblk(128);
    dim3 gridAh(N_ / 128, N_ / 64, B_);     // (2,4,8)
    dim3 gridZR(512 / 128, BN_ / 64, 1);    // (4,32)
    dim3 gridC (H_  / 128, BN_ / 64, 1);    // (2,32)
    dim3 gridGX(G3_ / 128, BN_ / 64, 1);    // (6,32)

    // -- per-call prep (deterministic): convert/transpose constants, zero state
    k_cvt_bf<<<(N_ * N_ + 255) / 256, blk, 0, stream>>>(sadj, A_bf, N_ * N_);
    k_transpose_bf<<<(H_ * G3_ + 255) / 256, blk, 0, stream>>>(Wh0, Wh0_t, H_, G3_);
    k_transpose_bf<<<(H_ * G3_ + 255) / 256, blk, 0, stream>>>(Wh1, Wh1_t, H_, G3_);
    k_transpose_bf<<<(H_ * G3_ + 255) / 256, blk, 0, stream>>>(Wx1, Wx1_t, H_, G3_);
    k_zero_f<<<(2 * BN_ * H_ + 255) / 256, blk, 0, stream>>>(h0f, 2 * BN_ * H_);          // h0f+h1f
    k_zero_f<<<(B_ * H_ * N_ + 255) / 256, blk, 0, stream>>>((float*)h0t, B_ * H_ * N_);  // h0t+h1t

    for (int t = 0; t < T_; ++t) {
        // ---- layer 0: gx = (A @ x_t) @ Wx0 + b0 (DIN=6, VALU)
        k_ax0<<<(B_ * N_ * DIN_ + 255) / 256, blk, 0, stream>>>(x2d, mask, sadj, ax0, t);
        k_gx0<<<(BN_ * G3_ + 255) / 256, blk, 0, stream>>>(ax0, Wx0, b0, gx);
        // ---- layer 0 recurrence
        k_gemm_bf_out<<<gridAh, gblk, 0, stream>>>(A_bf, h0t, Ah_bf);                 // Ah = A@h0
        k_gemm_zr<<<gridZR, gblk, 0, stream>>>(Ah_bf, Wh0_t, gx, h0f, zbuf, rht);     // z, r*h
        k_gemm_bf_out<<<gridAh, gblk, 0, stream>>>(A_bf, rht, Ah_bf);                 // Arh = A@(r*h)
        k_gemm_cupd<<<gridC, gblk, 0, stream>>>(Ah_bf, Wh0_t + 512 * H_, gx, zbuf, h0f, h0t);

        // ---- layer 1: gx = (A @ h0_t) @ Wx1 + b1 (WMMA)
        k_gemm_bf_out<<<gridAh, gblk, 0, stream>>>(A_bf, h0t, Axt_bf);                // A@h0_t
        k_gemm_gx<<<gridGX, gblk, 0, stream>>>(Axt_bf, Wx1_t, b1, gx);
        // ---- layer 1 recurrence
        k_gemm_bf_out<<<gridAh, gblk, 0, stream>>>(A_bf, h1t, Ah_bf);
        k_gemm_zr<<<gridZR, gblk, 0, stream>>>(Ah_bf, Wh1_t, gx, h1f, zbuf, rht);
        k_gemm_bf_out<<<gridAh, gblk, 0, stream>>>(A_bf, rht, Ah_bf);
        k_gemm_cupd<<<gridC, gblk, 0, stream>>>(Ah_bf, Wh1_t + 512 * H_, gx, zbuf, h1f, h1t);

        // ---- output head (DOUT=9, VALU)
        k_out<<<(B_ * N_ * DOUT_ + 255) / 256, blk, 0, stream>>>(h1f, Wout, bout, out, t);
    }
}
